// LatentODESurvival_40166534152317
// MI455X (gfx1250) — matrix operand order repaired and checked by reading, not compiled, not run on hardware
//
#include <hip/hip_runtime.h>
#include <math.h>

// Problem dims
#define BATCH 4096
#define TLEN  48
#define DIN   32
#define HGRU  256
#define LDIM  128
#define HID   256
#define EDIM  8
#define NHOR  48

typedef __attribute__((ext_vector_type(16))) __bf16 bf16x16;
typedef __attribute__((ext_vector_type(8)))  float  f32x8;

union FragAB { int i[8]; bf16x16 v; };

__device__ __forceinline__ unsigned short f2bf(float x) {
  unsigned int u = __float_as_uint(x);
  u += 0x7FFFu + ((u >> 16) & 1u);   // round-to-nearest-even
  return (unsigned short)(u >> 16);
}
__device__ __forceinline__ float sigf(float x) { return 1.f / (1.f + expf(-x)); }

// ---------------------------------------------------------------------------
// Generic bf16 WMMA GEMM:  C[M,N] = A[M,K] * Bpacked[K,N] (+bias)(+act)
// A: row-major bf16, K % 32 == 0.
// Bp: fragment-packed [N/16][K/32][32 lanes][16 bf16].
// Wave computes MI*16 x NI*16; 8 waves arranged WM x WN.
// Block tile = (WM*MI*16) x (WN*NI*16); M,N must divide block tile.
// act: 0 = none, 1 = tanh.  Cf (f32) and Cbf (bf16) each optional.
// ---------------------------------------------------------------------------
template <int MI, int NI, int WM, int WN>
__global__ __launch_bounds__(256)
void wmma_gemm(const unsigned short* __restrict__ A,
               const unsigned short* __restrict__ Bp,
               float* __restrict__ Cf,
               unsigned short* __restrict__ Cbf,
               const float* __restrict__ bias,
               int M, int N, int K, int act)
{
  const int tid  = threadIdx.x;
  const int wave = tid >> 5, lane = tid & 31;
  const int r    = lane & 15, half = lane >> 4;
  const int waveM = wave % WM, waveN = wave / WM;
  const int m0 = blockIdx.y * (WM * MI * 16) + waveM * (MI * 16);
  const int n0 = blockIdx.x * (WN * NI * 16) + waveN * (NI * 16);

  f32x8 acc[MI][NI];
#pragma unroll
  for (int i = 0; i < MI; ++i)
#pragma unroll
    for (int j = 0; j < NI; ++j)
#pragma unroll
      for (int e = 0; e < 8; ++e) acc[i][j][e] = 0.f;

  const int ktiles = K >> 5;
  for (int kt = 0; kt < ktiles; ++kt) {
    FragAB fa[MI], fb[NI];
#pragma unroll
    for (int mi = 0; mi < MI; ++mi) {
      const unsigned short* ar = A + (size_t)(m0 + mi * 16 + r) * K + kt * 32;
#pragma unroll
      for (int v = 0; v < 8; ++v) {
        const int kofs = half * 16 + 2 * v;
        fa[mi].i[v] = *(const int*)(ar + kofs);
      }
    }
#pragma unroll
    for (int ni = 0; ni < NI; ++ni) {
      const int* br = (const int*)(Bp + ((((size_t)(n0 >> 4) + ni) * ktiles + kt) * 32 + lane) * 16);
#pragma unroll
      for (int v = 0; v < 8; ++v) fb[ni].i[v] = br[v];
    }
#pragma unroll
    for (int mi = 0; mi < MI; ++mi)
#pragma unroll
      for (int ni = 0; ni < NI; ++ni)
        acc[mi][ni] = __builtin_amdgcn_wmma_f32_16x16x32_bf16(
            false, fa[mi].v, false, fb[ni].v, (short)0, acc[mi][ni], false, false);
  }

#pragma unroll
  for (int mi = 0; mi < MI; ++mi)
#pragma unroll
    for (int ni = 0; ni < NI; ++ni) {
      const int col = n0 + ni * 16 + r;
      const float bv = bias ? bias[col] : 0.f;
#pragma unroll
      for (int i = 0; i < 8; ++i) {
        const int row = m0 + mi * 16 + half * 8 + i;
        float val = acc[mi][ni][i] + bv;
        if (act == 1) val = tanhf(val);
        const size_t off = (size_t)row * N + col;
        if (Cf)  Cf[off]  = val;
        if (Cbf) Cbf[off] = f2bf(val);
      }
    }
}

// ---------------------------------------------------------------------------
// Weight packing into WMMA B-fragment layout.
// B[k,n] = W[n,k] (torch-style [out,in] weight). Zero-padded to Npad/Kpad.
// Layout: Bp[((nt*Ktiles + kt)*32 + lane)*16 + e], lane: n = nt*16 + lane%16,
// half = lane/16, element e: v=e/2, k = kt*32 + 16*half + 2v + (e&1).
// ---------------------------------------------------------------------------
__global__ void pack_b(const float* __restrict__ W, unsigned short* __restrict__ Bp,
                       int Nsrc, int Ksrc, int Npad, int Kpad)
{
  const int total = Npad * Kpad;
  const int idx = blockIdx.x * blockDim.x + threadIdx.x;
  if (idx >= total) return;
  const int e = idx & 15, lane = (idx >> 4) & 31, rest = idx >> 9;
  const int ktiles = Kpad >> 5;
  const int kt = rest % ktiles, nt = rest / ktiles;
  const int n = nt * 16 + (lane & 15);
  const int k = kt * 32 + 16 * (lane >> 4) + 2 * (e >> 1) + (e & 1);
  const float v = (n < Nsrc && k < Ksrc) ? W[n * Ksrc + k] : 0.f;
  Bp[idx] = f2bf(v);
}

// Combined GRU weight: K = 320 (h:0..255 | x:256..287 | mask:288..319),
// N = 1024 (0..511: r,z gates w_hh+w_ih | 512..767: i_n (x only) | 768..1023: h_n (h only))
__device__ __forceinline__ float gru_wc(int n, int k, const float* w_ih, const float* w_hh) {
  if (n < 512)  return (k < 256) ? w_hh[n * 256 + k] : w_ih[n * 64 + (k - 256)];
  if (n < 768)  return (k < 256) ? 0.f : w_ih[n * 64 + (k - 256)];          // i_n rows 512..767
  return (k < 256) ? w_hh[(n - 256) * 256 + k] : 0.f;                       // h_n rows 512..767
}

__global__ void pack_gru_wc(const float* __restrict__ w_ih, const float* __restrict__ w_hh,
                            unsigned short* __restrict__ Bp)
{
  const int total = 1024 * 320;
  const int idx = blockIdx.x * blockDim.x + threadIdx.x;
  if (idx >= total) return;
  const int e = idx & 15, lane = (idx >> 4) & 31, rest = idx >> 9;
  const int ktiles = 320 >> 5; // 10
  const int kt = rest % ktiles, nt = rest / ktiles;
  const int n = nt * 16 + (lane & 15);
  const int k = kt * 32 + 16 * (lane >> 4) + 2 * (e >> 1) + (e & 1);
  Bp[idx] = f2bf(gru_wc(n, k, w_ih, w_hh));
}

// ---------------------------------------------------------------------------
__global__ void any_obs_kernel(const float* __restrict__ mask, float* __restrict__ obs)
{
  const int idx = blockIdx.x * blockDim.x + threadIdx.x; // over B*T
  if (idx >= BATCH * TLEN) return;
  float s = 0.f;
  const float* m = mask + (size_t)idx * DIN;
  for (int d = 0; d < DIN; ++d) s += m[d];
  obs[idx] = (s > 0.f) ? 1.f : 0.f;
}

__global__ void zero_f32(float* __restrict__ p, int n)
{
  const int idx = blockIdx.x * blockDim.x + threadIdx.x;
  if (idx < n) p[idx] = 0.f;
}

// A for the GRU step GEMM: [4096, 320] bf16 = [h | X_t | mask_t]
__global__ void pack_a_gru(const float* __restrict__ h, const float* __restrict__ X,
                           const float* __restrict__ mask, unsigned short* __restrict__ A, int t)
{
  const int idx = blockIdx.x * blockDim.x + threadIdx.x;
  if (idx >= BATCH * 320) return;
  const int b = idx / 320, c = idx % 320;
  float v;
  if (c < 256)      v = h[b * 256 + c];
  else if (c < 288) v = X[((size_t)b * TLEN + t) * DIN + (c - 256)];
  else              v = mask[((size_t)b * TLEN + t) * DIN + (c - 288)];
  A[idx] = f2bf(v);
}

// GRU cell pointwise math from fused gate GEMM output G[4096,1024]
__global__ void gru_gate(const float* __restrict__ G, const float* __restrict__ b_ih,
                         const float* __restrict__ b_hh, const float* __restrict__ obs,
                         float* __restrict__ h, int t)
{
  const int idx = blockIdx.x * blockDim.x + threadIdx.x;
  if (idx >= BATCH * HGRU) return;
  const int b = idx / HGRU, j = idx % HGRU;
  const float* g = G + (size_t)b * 1024;
  const float rpre = g[j]       + b_ih[j]       + b_hh[j];
  const float zpre = g[256 + j] + b_ih[256 + j] + b_hh[256 + j];
  const float i_n  = g[512 + j] + b_ih[512 + j];
  const float h_n  = g[768 + j] + b_hh[512 + j];
  const float rr = sigf(rpre), zg = sigf(zpre);
  const float nn = tanhf(i_n + rr * h_n);
  const float ho = h[idx];
  const float hn = (1.f - zg) * nn + zg * ho;
  const float o  = obs[b * TLEN + t];
  h[idx] = o * hn + (1.f - o) * ho;
}

__global__ void f32_to_bf16(const float* __restrict__ src, unsigned short* __restrict__ dst, int n)
{
  const int idx = blockIdx.x * blockDim.x + threadIdx.x;
  if (idx < n) dst[idx] = f2bf(src[idx]);
}

// split z0out -> (mean -> d_out, logvar -> d_out, z state)
__global__ void split_z0(const float* __restrict__ z0out, float* __restrict__ mean_out,
                         float* __restrict__ logvar_out, float* __restrict__ z)
{
  const int idx = blockIdx.x * blockDim.x + threadIdx.x;
  if (idx >= BATCH * LDIM) return;
  const int b = idx / LDIM, j = idx % LDIM;
  const float m  = z0out[(size_t)b * 256 + j];
  const float lv = z0out[(size_t)b * 256 + 128 + j];
  mean_out[idx] = m;
  logvar_out[idx] = lv;
  z[idx] = m;
}

// Sinusoidal time embedding projected: temb[8] (shared by all batch rows)
__global__ void temb_kernel(const float* __restrict__ tproj_w, const float* __restrict__ tproj_b,
                            float* __restrict__ temb, float t)
{
  const int j = threadIdx.x;
  if (j >= EDIM) return;
  const float PI = 3.14159265358979323846f;
  float acc = tproj_b[j];
  for (int i = 0; i < EDIM; ++i) {
    const float emb = (i < 4) ? sinf(t * (float)(i + 1) * PI) : cosf(t * (float)(i - 3) * PI);
    acc += tproj_w[j * EDIM + i] * emb;
  }
  temb[j] = acc;
}

// A for ODE MLP layer 1: [4096, 160] bf16 = [z + coef*k | temb | zero pad]
__global__ void pack_a_ode(const float* __restrict__ z, const float* __restrict__ k,
                           float coef, const float* __restrict__ temb,
                           unsigned short* __restrict__ A)
{
  const int idx = blockIdx.x * blockDim.x + threadIdx.x;
  if (idx >= BATCH * 160) return;
  const int b = idx / 160, c = idx % 160;
  float v;
  if (c < 128)      v = z[b * 128 + c] + (k ? coef * k[b * 128 + c] : 0.f);
  else if (c < 136) v = temb[c - 128];
  else              v = 0.f;
  A[idx] = f2bf(v);
}

__global__ void rk4_update(float* __restrict__ z, const float* __restrict__ k1,
                           const float* __restrict__ k2, const float* __restrict__ k3,
                           const float* __restrict__ k4, float h6)
{
  const int idx = blockIdx.x * blockDim.x + threadIdx.x;
  if (idx >= BATCH * LDIM) return;
  z[idx] += h6 * (k1[idx] + 2.f * k2[idx] + 2.f * k3[idx] + k4[idx]);
}

// DeepHit head: one wave per batch row (wave32), 32 lanes = 32 hidden units
__global__ __launch_bounds__(256)
void head_kernel(const float* __restrict__ z, const float* __restrict__ w1,
                 const float* __restrict__ b1, const float* __restrict__ w2,
                 const float* __restrict__ b2, float* __restrict__ hazard, int ti)
{
  const int wave = threadIdx.x >> 5, lane = threadIdx.x & 31;
  const int b = blockIdx.x * 8 + wave;
  const float* zr = z + (size_t)b * LDIM;
  const float* wr = w1 + lane * LDIM;
  float acc = 0.f;
  for (int d = 0; d < LDIM; ++d) acc += zr[d] * wr[d];
  float contrib = fmaxf(acc + b1[lane], 0.f) * w2[lane];
  for (int off = 16; off > 0; off >>= 1) contrib += __shfl_xor(contrib, off, 32);
  if (lane == 0) hazard[(size_t)b * NHOR + ti] = sigf(contrib + b2[0]);
}

__global__ void surv_kernel(const float* __restrict__ hazard, float* __restrict__ survival,
                            float* __restrict__ pg)
{
  const int b = blockIdx.x * blockDim.x + threadIdx.x;
  if (b >= BATCH) return;
  float ls = 0.f, last = 1.f;
  for (int i = 0; i < NHOR; ++i) {
    const float s = expf(ls);
    survival[(size_t)b * NHOR + i] = s;
    last = s;
    ls += logf(1.f - hazard[(size_t)b * NHOR + i] + 1e-7f);
  }
  pg[b] = 1.f - last;
}

// ---------------------------------------------------------------------------
static inline int ceil_div(int a, int b) { return (a + b - 1) / b; }

extern "C" void kernel_launch(void* const* d_in, const int* in_sizes, int n_in,
                              void* d_out, int out_size, void* d_ws, size_t ws_size,
                              hipStream_t stream)
{
  const float* X        = (const float*)d_in[0];
  const float* mask     = (const float*)d_in[1];
  const float* gru_w_ih = (const float*)d_in[2];
  const float* gru_w_hh = (const float*)d_in[3];
  const float* gru_b_ih = (const float*)d_in[4];
  const float* gru_b_hh = (const float*)d_in[5];
  const float* z0_w     = (const float*)d_in[6];
  const float* z0_b     = (const float*)d_in[7];
  const float* tproj_w  = (const float*)d_in[8];
  const float* tproj_b  = (const float*)d_in[9];
  const float* ode_w1   = (const float*)d_in[10];
  const float* ode_b1   = (const float*)d_in[11];
  const float* ode_w2   = (const float*)d_in[12];
  const float* ode_b2   = (const float*)d_in[13];
  const float* ode_w3   = (const float*)d_in[14];
  const float* ode_b3   = (const float*)d_in[15];
  const float* sh_w1    = (const float*)d_in[16];
  const float* sh_b1    = (const float*)d_in[17];
  const float* sh_w2    = (const float*)d_in[18];
  const float* sh_b2    = (const float*)d_in[19];

  float* out = (float*)d_out;
  float* out_hazard = out;                       // [B, NH]
  float* out_surv   = out + BATCH * NHOR;        // [B, NH]
  float* out_mean   = out + 2 * BATCH * NHOR;    // [B, L]
  float* out_logv   = out_mean + BATCH * LDIM;   // [B, L]
  float* out_pg     = out_logv + BATCH * LDIM;   // [B]

  // ---- workspace carve-out ----
  char* ws = (char*)d_ws;
  auto take = [&](size_t bytes) { void* p = ws; ws += (bytes + 255) & ~(size_t)255; return p; };
  unsigned short* bp_wc  = (unsigned short*)take(1024 * 320 * 2);
  unsigned short* bp_z0  = (unsigned short*)take(256 * 256 * 2);
  unsigned short* bp_w1  = (unsigned short*)take(256 * 160 * 2);
  unsigned short* bp_w2  = (unsigned short*)take(256 * 256 * 2);
  unsigned short* bp_w3  = (unsigned short*)take(128 * 256 * 2);
  float*          obs    = (float*)take((size_t)BATCH * TLEN * 4);
  float*          hbuf   = (float*)take((size_t)BATCH * HGRU * 4);
  unsigned short* Agru   = (unsigned short*)take((size_t)BATCH * 320 * 2);
  float*          Gbuf   = (float*)take((size_t)BATCH * 1024 * 4);
  unsigned short* Az     = (unsigned short*)take((size_t)BATCH * 256 * 2);
  float*          z0out  = (float*)take((size_t)BATCH * 256 * 4);
  float*          zbuf   = (float*)take((size_t)BATCH * LDIM * 4);
  float*          temb   = (float*)take(EDIM * 4);
  unsigned short* Aode   = (unsigned short*)take((size_t)BATCH * 160 * 2);
  unsigned short* h1bf   = (unsigned short*)take((size_t)BATCH * HID * 2);
  unsigned short* h2bf   = (unsigned short*)take((size_t)BATCH * HID * 2);
  float*          kst[4];
  for (int i = 0; i < 4; ++i) kst[i] = (float*)take((size_t)BATCH * LDIM * 4);

  const dim3 blk(256);

  // ---- weight packs (deterministic, every call) ----
  pack_gru_wc<<<ceil_div(1024 * 320, 256), blk, 0, stream>>>(gru_w_ih, gru_w_hh, bp_wc);
  pack_b<<<ceil_div(256 * 256, 256), blk, 0, stream>>>(z0_w,  bp_z0, 256, 256, 256, 256);
  pack_b<<<ceil_div(256 * 160, 256), blk, 0, stream>>>(ode_w1, bp_w1, 256, 136, 256, 160);
  pack_b<<<ceil_div(256 * 256, 256), blk, 0, stream>>>(ode_w2, bp_w2, 256, 256, 256, 256);
  pack_b<<<ceil_div(128 * 256, 256), blk, 0, stream>>>(ode_w3, bp_w3, 128, 256, 128, 256);

  any_obs_kernel<<<ceil_div(BATCH * TLEN, 256), blk, 0, stream>>>(mask, obs);
  zero_f32<<<ceil_div(BATCH * HGRU, 256), blk, 0, stream>>>(hbuf, BATCH * HGRU);

  // ---- reversed-time GRU scan: one fused WMMA GEMM per step ----
  // Big tile config: 64x64 per wave (4x4 frags), 8 waves as 4x2 -> 256x128 block.
  // Arithmetic intensity ~32 FLOP per L2 byte (vs 16 with 2x2): the recurrent
  // GEMM is load-bound, so reuse in registers is what MI455X rewards here.
  for (int t = TLEN - 1; t >= 0; --t) {
    pack_a_gru<<<ceil_div(BATCH * 320, 256), blk, 0, stream>>>(hbuf, X, mask, Agru, t);
    wmma_gemm<4, 4, 4, 2><<<dim3(1024 / 128, BATCH / 256), blk, 0, stream>>>(
        Agru, bp_wc, Gbuf, nullptr, nullptr, BATCH, 1024, 320, 0);
    gru_gate<<<ceil_div(BATCH * HGRU, 256), blk, 0, stream>>>(Gbuf, gru_b_ih, gru_b_hh, obs, hbuf, t);
  }

  // ---- z0 projection (small config: more blocks for occupancy) ----
  f32_to_bf16<<<ceil_div(BATCH * 256, 256), blk, 0, stream>>>(hbuf, Az, BATCH * 256);
  wmma_gemm<2, 2, 4, 2><<<dim3(256 / 64, BATCH / 128), blk, 0, stream>>>(
      Az, bp_z0, z0out, nullptr, z0_b, BATCH, 256, 256, 0);
  split_z0<<<ceil_div(BATCH * LDIM, 256), blk, 0, stream>>>(z0out, out_mean, out_logv, zbuf);

  // ---- hazard at t_pts[0] ----
  head_kernel<<<BATCH / 8, blk, 0, stream>>>(zbuf, sh_w1, sh_b1, sh_w2, sh_b2, out_hazard, 0);

  // ---- fixed-step RK4 over the 47 intervals of linspace(0,1,48) ----
  const float hstep = 1.f / 47.f;
  auto ode_eval = [&](float t, const float* kin, float coef, float* kout) {
    temb_kernel<<<1, 32, 0, stream>>>(tproj_w, tproj_b, temb, t);
    pack_a_ode<<<ceil_div(BATCH * 160, 256), blk, 0, stream>>>(zbuf, kin, coef, temb, Aode);
    wmma_gemm<2, 2, 4, 2><<<dim3(256 / 64, BATCH / 128), blk, 0, stream>>>(
        Aode, bp_w1, nullptr, h1bf, ode_b1, BATCH, 256, 160, 1);
    wmma_gemm<2, 2, 4, 2><<<dim3(256 / 64, BATCH / 128), blk, 0, stream>>>(
        h1bf, bp_w2, nullptr, h2bf, ode_b2, BATCH, 256, 256, 1);
    wmma_gemm<2, 2, 4, 2><<<dim3(128 / 64, BATCH / 128), blk, 0, stream>>>(
        h2bf, bp_w3, kout, nullptr, ode_b3, BATCH, 128, 256, 0);
  };
  // stages 2 & 3 share the same time point; temb is just recomputed (cheap, deterministic)
  for (int s = 0; s < NHOR - 1; ++s) {
    const float t0 = (float)s * hstep;
    ode_eval(t0,                nullptr, 0.f,          kst[0]);
    ode_eval(t0 + 0.5f * hstep, kst[0],  0.5f * hstep, kst[1]);
    ode_eval(t0 + 0.5f * hstep, kst[1],  0.5f * hstep, kst[2]);
    ode_eval(t0 + hstep,        kst[2],  hstep,        kst[3]);
    rk4_update<<<ceil_div(BATCH * LDIM, 256), blk, 0, stream>>>(
        zbuf, kst[0], kst[1], kst[2], kst[3], hstep / 6.f);
    head_kernel<<<BATCH / 8, blk, 0, stream>>>(zbuf, sh_w1, sh_b1, sh_w2, sh_b2, out_hazard, s + 1);
  }

  // ---- survival curve + p_global ----
  surv_kernel<<<ceil_div(BATCH, 256), blk, 0, stream>>>(out_hazard, out_surv, out_pg);

  (void)in_sizes; (void)n_in; (void)out_size; (void)ws_size;
}
